// Upsampler3DWithAttention_60670708023496
// MI455X (gfx1250) — matrix operand order repaired and checked by reading, not compiled, not run on hardware
//
#include <hip/hip_runtime.h>

// ---------------------------------------------------------------------------
// Upsampler (1x1x1 Conv3d == GEMM) + fused 3D pixel shuffle for gfx1250.
// GEMM: M=8192 (voxels), K=1024 (in ch), N=3584 (out ch = 14*4*8*8).
// fp32 inputs are split into bf16 hi/lo pairs; each C tile accumulates
// hi*hi + hi*lo + lo*hi via v_wmma_f32_16x16x32_bf16 (f32 accumulate),
// giving ~fp32 accuracy at BF16 matrix-pipe rate.
// ---------------------------------------------------------------------------

typedef __attribute__((ext_vector_type(16))) __bf16        v16bf;
typedef __attribute__((ext_vector_type(8)))  float         v8f;
typedef __attribute__((ext_vector_type(8)))  unsigned short u16x8;
typedef __attribute__((ext_vector_type(16))) unsigned short u16x16;
typedef __attribute__((ext_vector_type(4)))  float         f32x4;

#define K_DIM 1024
#define N_DIM 3584
#define M_DIM 8192
#define MB 64     // block tile M
#define NB 128    // block tile N
#define KB 32     // K step (= bf16 WMMA depth)

// Split fp32 pair -> packed bf16 hi words and bf16 lo (residual) words.
__device__ __forceinline__ void split2(float f0, float f1,
                                       unsigned int& hi, unsigned int& lo) {
  unsigned int u0 = __float_as_uint(f0);
  unsigned int u1 = __float_as_uint(f1);
  float h0 = __uint_as_float(u0 & 0xFFFF0000u);   // truncated bf16 (exact)
  float h1 = __uint_as_float(u1 & 0xFFFF0000u);
  float r0 = f0 - h0;                              // exact residual
  float r1 = f1 - h1;
  hi = (u0 >> 16) | (u1 & 0xFFFF0000u);
  lo = (__float_as_uint(r0) >> 16) | (__float_as_uint(r1) & 0xFFFF0000u);
}

__global__ __launch_bounds__(256)
void upsample3d_gemm_shuffle(const float* __restrict__ X,
                             const float* __restrict__ Wm,
                             const float* __restrict__ Bias,
                             float* __restrict__ Out) {
  // LDS staging: A row-major [m][k], B transposed to [n][k] so each lane's
  // B fragment (16 consecutive K for its column) is one contiguous 32B read.
  __shared__ __align__(16) unsigned short AhiS[MB][KB];  // 4 KB
  __shared__ __align__(16) unsigned short AloS[MB][KB];  // 4 KB
  __shared__ __align__(16) unsigned short BhiS[NB][KB];  // 8 KB
  __shared__ __align__(16) unsigned short BloS[NB][KB];  // 8 KB

  const int tid   = threadIdx.x;
  const int lane  = tid & 31;           // wave32
  const int wave  = tid >> 5;           // 8 waves
  const int half  = lane >> 4;          // K-half selector (A/B fragments)
  const int r16   = lane & 15;
  const int mBase = blockIdx.y * MB;
  const int nBase = blockIdx.x * NB;
  const int nWave = nBase + wave * 16;  // each wave owns a 16-wide N strip

  v8f acc[4];
#pragma unroll
  for (int i = 0; i < 4; ++i)
#pragma unroll
    for (int j = 0; j < 8; ++j) acc[i][j] = 0.0f;

  // Cooperative tile-load assignments (coalesced b128 global loads).
  const int rowA = tid >> 2;            // 64 rows, 4 threads/row
  const int colA = (tid & 3) * 8;       // 8 floats each
  const int rowB = tid >> 3;            // 32 k-rows, 8 threads/row
  const int colB = (tid & 7) * 16;      // 16 floats each

  const float* pA = X  + (long)(mBase + rowA) * K_DIM + colA;
  const float* pB = Wm + (long)rowB * N_DIM + nBase + colB;

  for (int kt = 0; kt < K_DIM; kt += KB) {
    if (kt + KB < K_DIM) {              // uniform branch; global_prefetch_b8
      __builtin_prefetch(pA + kt + KB, 0, 3);
      __builtin_prefetch(pB + (long)(kt + KB) * N_DIM, 0, 3);
    }

    // ---- stage A (64x32 fp32 -> bf16 hi/lo, row-major) ----
    {
      const f32x4* p = (const f32x4*)(pA + kt);
      f32x4 v0 = p[0], v1 = p[1];
      float fv[8];
#pragma unroll
      for (int j = 0; j < 4; ++j) { fv[j] = v0[j]; fv[4 + j] = v1[j]; }
      unsigned int* dh = (unsigned int*)&AhiS[rowA][colA];
      unsigned int* dl = (unsigned int*)&AloS[rowA][colA];
#pragma unroll
      for (int j = 0; j < 4; ++j) {
        unsigned int hi, lo;
        split2(fv[2 * j], fv[2 * j + 1], hi, lo);
        dh[j] = hi; dl[j] = lo;
      }
    }

    // ---- stage B (32x128 fp32 -> bf16 hi/lo, transposed to [n][k]) ----
    {
      const f32x4* q = (const f32x4*)(pB + (long)kt * N_DIM);
      f32x4 b0 = q[0], b1 = q[1], b2 = q[2], b3 = q[3];
      float bv[16];
#pragma unroll
      for (int j = 0; j < 4; ++j) {
        bv[j] = b0[j]; bv[4 + j] = b1[j]; bv[8 + j] = b2[j]; bv[12 + j] = b3[j];
      }
#pragma unroll
      for (int j = 0; j < 16; ++j) {
        unsigned int u = __float_as_uint(bv[j]);
        float h = __uint_as_float(u & 0xFFFF0000u);
        float r = bv[j] - h;
        BhiS[colB + j][rowB] = (unsigned short)(u >> 16);
        BloS[colB + j][rowB] = (unsigned short)(__float_as_uint(r) >> 16);
      }
    }

    __syncthreads();

    // ---- compute: 4 M-tiles x (hi*hi + hi*lo + lo*hi) ----
    // B fragment: column n = lane%16, K rows (half?16..31:0..15) contiguous.
    u16x16 bhU = *(const u16x16*)&BhiS[wave * 16 + r16][half * 16];
    u16x16 blU = *(const u16x16*)&BloS[wave * 16 + r16][half * 16];
    v16bf bh = __builtin_bit_cast(v16bf, bhU);
    v16bf bl = __builtin_bit_cast(v16bf, blU);

#pragma unroll
    for (int mt = 0; mt < 4; ++mt) {
      const int row = mt * 16 + r16;    // A row M = lane%16
      // A fragment: elems0-15 = K(half?16..31:0..15) contiguous.
      u16x8 a0 = *(const u16x8*)&AhiS[row][half * 16];
      u16x8 a1 = *(const u16x8*)&AhiS[row][half * 16 + 8];
      u16x8 a2 = *(const u16x8*)&AloS[row][half * 16];
      u16x8 a3 = *(const u16x8*)&AloS[row][half * 16 + 8];
      u16x16 ahU = __builtin_shufflevector(a0, a1, 0,1,2,3,4,5,6,7,8,9,10,11,12,13,14,15);
      u16x16 alU = __builtin_shufflevector(a2, a3, 0,1,2,3,4,5,6,7,8,9,10,11,12,13,14,15);
      v16bf ah = __builtin_bit_cast(v16bf, ahU);
      v16bf al = __builtin_bit_cast(v16bf, alU);

      acc[mt] = __builtin_amdgcn_wmma_f32_16x16x32_bf16(
          false, ah, false, bh, (short)0, acc[mt], false, false);
      acc[mt] = __builtin_amdgcn_wmma_f32_16x16x32_bf16(
          false, ah, false, bl, (short)0, acc[mt], false, false);
      acc[mt] = __builtin_amdgcn_wmma_f32_16x16x32_bf16(
          false, al, false, bh, (short)0, acc[mt], false, false);
    }

    __syncthreads();
  }

  // ---- epilogue: bias + fused 3D pixel-shuffle scatter ----
  // C tile layout: N = lane%16 (all vgprs), M = mBase + mt*16 + 8*half + i.
  const int nGlob = nWave + r16;        // output channel d in [0,3584)
  const float bias = Bias[nGlob];
  const int c  = nGlob >> 8;            // d = c*256 + rt*64 + rh*8 + rw
  const int rem = nGlob & 255;
  const int rt = rem >> 6;
  const int rh = (rem >> 3) & 7;
  const int rw = rem & 7;

#pragma unroll
  for (int mt = 0; mt < 4; ++mt) {
    const int m0 = mBase + mt * 16 + half * 8;  // 8 consecutive voxels
    const int w0 = m0 & 31;                     // no 32-boundary crossing
    const int h  = (m0 >> 5) & 31;
    const int t  = m0 >> 10;
    // out[(t*4+rt), (h*8+rh), (w*8+rw), c], shape (32,256,256,14)
    const int base =
        (((t * 4 + rt) * 256 + h * 8 + rh) * 256 + w0 * 8 + rw) * 14 + c;
#pragma unroll
    for (int i = 0; i < 8; ++i)
      Out[base + i * (8 * 14)] = acc[mt][i] + bias;  // w+1 -> +8*14 floats
  }
}

extern "C" void kernel_launch(void* const* d_in, const int* in_sizes, int n_in,
                              void* d_out, int out_size, void* d_ws, size_t ws_size,
                              hipStream_t stream) {
  (void)in_sizes; (void)n_in; (void)out_size; (void)d_ws; (void)ws_size;
  const float* x  = (const float*)d_in[0];  // (1,8,32,32,1024) fp32
  const float* wm = (const float*)d_in[1];  // (1024,3584) fp32
  const float* b  = (const float*)d_in[2];  // (3584,) fp32
  float* out = (float*)d_out;               // (1,32,256,256,14) fp32

  dim3 grid(N_DIM / NB, M_DIM / MB);        // 28 x 128
  dim3 block(256);                          // 8 waves (wave32)
  upsample3d_gemm_shuffle<<<grid, block, 0, stream>>>(x, wm, b, out);
}
